// MPN_35158602285571
// MI455X (gfx1250) — matrix operand order repaired and checked by reading, not compiled, not run on hardware
//
#include <hip/hip_runtime.h>
#include <hip/hip_bf16.h>

#define N_ATOMS   100000
#define N_BONDS   200001
#define ATOM_FDIM 39
#define BOND_FDIM 11
#define BOND_IN   (ATOM_FDIM + BOND_FDIM)   // 50
#define MAX_NB    6
#define HIDDEN    256
#define DEPTH     6
#define N_MOLS    2000
#define APM       (N_ATOMS / N_MOLS)        // 50
#define WO_COLS   (ATOM_FDIM + HIDDEN)      // 295

#define M_TILE    32
#define USE_ASYNC_BIN 1

typedef __bf16 bf16_t;
typedef __attribute__((ext_vector_type(16))) __bf16 v16bf;
typedef __attribute__((ext_vector_type(8)))  __bf16 v8bf;
typedef __attribute__((ext_vector_type(8)))  float  v8f;

#define A_STRIDE 280   // 140 dwords/row -> 12*m mod 64 distinct (conflict-free b128), 16B aligned

static __device__ __forceinline__ v16bf combine16(v8bf lo, v8bf hi) {
    v16bf r;
#pragma unroll
    for (int i = 0; i < 8; ++i) { r[i] = lo[i]; r[i + 8] = hi[i]; }
    return r;
}

// ---------------------------------------------------------------- weights -> bf16
__global__ void k_convert(const float* __restrict__ Wh, const float* __restrict__ Wo,
                          bf16_t* __restrict__ Whb, bf16_t* __restrict__ Wob) {
    int idx = blockIdx.x * 256 + threadIdx.x;   // 65536 total
    int h = idx >> 8, k = idx & 255;
    Whb[idx] = (bf16_t)Wh[idx];                         // Whb[h][k] = W_h[h][k]
    Wob[idx] = (bf16_t)Wo[h * WO_COLS + ATOM_FDIM + k]; // Wob[h][k] = W_o[h][39+k]
}

// ------------------------------------------------- binput = fbonds @ W_i^T ; msg0 = relu
__global__ void __launch_bounds__(256) k_init(const float* __restrict__ fb,
                                              const float* __restrict__ Wi,
                                              bf16_t* __restrict__ binb,
                                              bf16_t* __restrict__ msg0) {
    int b = blockIdx.x;          // one bond per block
    int h = threadIdx.x;         // one hidden channel per thread
    const float* fr = fb + (size_t)b * BOND_IN;
    const float* wr = Wi + (size_t)h * BOND_IN;
    float acc = 0.f;
#pragma unroll
    for (int k = 0; k < BOND_IN; ++k) acc = fmaf(fr[k], wr[k], acc);
    size_t o = (size_t)b * HIDDEN + h;
    binb[o] = (bf16_t)acc;
    msg0[o] = (bf16_t)fmaxf(acc, 0.f);
}

// ------------------------------------------- one message-passing round (gather + WMMA)
__global__ void __launch_bounds__(256) k_round(const bf16_t* __restrict__ msg_in,
                                               bf16_t* __restrict__ msg_out,
                                               const bf16_t* __restrict__ binb,
                                               const int* __restrict__ bgraph,
                                               const bf16_t* __restrict__ Whb) {
    __shared__ __align__(16) bf16_t At[M_TILE * A_STRIDE];
#if USE_ASYNC_BIN
    __shared__ __align__(16) bf16_t BinT[M_TILE * HIDDEN];   // 16 KB residual tile
#endif
    const int tid  = threadIdx.x;
    const int base = blockIdx.x * M_TILE;

#if USE_ASYNC_BIN
    // Stage the 32x256 bf16 binput tile into LDS with async-to-LDS DMA; overlaps the
    // whole gather + WMMA phase. Full EXEC here. Tail rows clamp to the last bond.
    {
        unsigned ldsBase = (unsigned)(uintptr_t)(void*)BinT;          // flat[31:0] == LDS offset
        unsigned long long sbase = (unsigned long long)(uintptr_t)binb;
#pragma unroll
        for (int it = 0; it < 4; ++it) {
            int off = (it * 256 + tid) * 16;                          // [0, 16384)
            int rb  = base + (off >> 9);
            if (rb > N_BONDS - 1) rb = N_BONDS - 1;
            unsigned voff = (unsigned)rb * 512u + (unsigned)(off & 511);
            unsigned ldsa = ldsBase + (unsigned)off;
            asm volatile("global_load_async_to_lds_b128 %0, %1, %2"
                         :: "v"(ldsa), "v"(voff), "s"(sbase) : "memory");
        }
    }
#endif

    // Phase A: gather-sum 6 neighbor messages into LDS A-tile (fp32 accumulate)
    {
        int mrow = tid >> 4;          // 0..15
        int ks   = (tid & 15) * 16;   // 16 channels per thread
#pragma unroll
        for (int half = 0; half < 2; ++half) {
            int m = mrow + half * 16;
            int bond = base + m;
            float acc[16];
#pragma unroll
            for (int i = 0; i < 16; ++i) acc[i] = 0.f;
            if (bond < N_BONDS) {
                const int* nb = bgraph + (size_t)bond * MAX_NB;
#pragma unroll
                for (int j = 0; j < MAX_NB; ++j) {
                    int row = nb[j];
                    const v8bf* p = (const v8bf*)(msg_in + (size_t)row * HIDDEN + ks);
                    v8bf lo = p[0], hi = p[1];
#pragma unroll
                    for (int i = 0; i < 8; ++i) {
                        acc[i]     += (float)lo[i];
                        acc[i + 8] += (float)hi[i];
                    }
                }
            }
            v8bf s0, s1;
#pragma unroll
            for (int i = 0; i < 8; ++i) { s0[i] = (bf16_t)acc[i]; s1[i] = (bf16_t)acc[i + 8]; }
            *(v8bf*)&At[m * A_STRIDE + ks]     = s0;
            *(v8bf*)&At[m * A_STRIDE + ks + 8] = s1;
        }
    }
    __syncthreads();

    // Phase B: per wave, 32(M) x 32(N) outputs = 4 WMMA tiles, K = 256 in 8 chunks of 32
    const int lane = tid & 31;
    const int wave = tid >> 5;            // 0..7 -> N slab [wave*32, wave*32+32)
    const int hsel = lane >> 4;           // 0/1 half-wave
    const int l15  = lane & 15;
    const int n0   = wave * 32 + l15;
    const int n1   = n0 + 16;

    v8f c00 = {}; v8f c01 = {}; v8f c10 = {}; v8f c11 = {};
#pragma unroll
    for (int kc = 0; kc < 8; ++kc) {
        // A fragments (contiguous K order, matching B): elems 0..15 = K kb..kb+15
        int kb = kc * 32 + hsel * 16;
        v16bf a0 = combine16(*(const v8bf*)&At[l15 * A_STRIDE + kb],
                             *(const v8bf*)&At[l15 * A_STRIDE + kb + 8]);
        v16bf a1 = combine16(*(const v8bf*)&At[(l15 + 16) * A_STRIDE + kb],
                             *(const v8bf*)&At[(l15 + 16) * A_STRIDE + kb + 8]);
        // B fragments: lane = column n, 16 contiguous K values = W_h row n (contiguous)
        int k0 = kc * 32 + hsel * 16;
        const v8bf* pb0 = (const v8bf*)(Whb + (size_t)n0 * HIDDEN + k0);
        v16bf b0 = combine16(pb0[0], pb0[1]);
        const v8bf* pb1 = (const v8bf*)(Whb + (size_t)n1 * HIDDEN + k0);
        v16bf b1 = combine16(pb1[0], pb1[1]);
        c00 = __builtin_amdgcn_wmma_f32_16x16x32_bf16(false, a0, false, b0, (short)0, c00, false, false);
        c01 = __builtin_amdgcn_wmma_f32_16x16x32_bf16(false, a0, false, b1, (short)0, c01, false, false);
        c10 = __builtin_amdgcn_wmma_f32_16x16x32_bf16(false, a1, false, b0, (short)0, c10, false, false);
        c11 = __builtin_amdgcn_wmma_f32_16x16x32_bf16(false, a1, false, b1, (short)0, c11, false, false);
    }

#if USE_ASYNC_BIN
    asm volatile("s_wait_asynccnt 0x0" ::: "memory");   // own wave's DMA done
    __syncthreads();                                    // -> everyone's DMA done
#endif

    // Epilogue: residual + ReLU, store bf16. C layout: VGPR r -> M = hsel*8+r (+16), N = n0/n1
#pragma unroll
    for (int r = 0; r < 8; ++r) {
        int m = hsel * 8 + r;
        int bond0 = base + m;
        int bond1 = bond0 + 16;
        if (bond0 < N_BONDS) {
#if USE_ASYNC_BIN
            float r0 = (float)BinT[m * HIDDEN + n0];
            float r1 = (float)BinT[m * HIDDEN + n1];
#else
            float r0 = (float)binb[(size_t)bond0 * HIDDEN + n0];
            float r1 = (float)binb[(size_t)bond0 * HIDDEN + n1];
#endif
            msg_out[(size_t)bond0 * HIDDEN + n0] = (bf16_t)fmaxf(c00[r] + r0, 0.f);
            msg_out[(size_t)bond0 * HIDDEN + n1] = (bf16_t)fmaxf(c01[r] + r1, 0.f);
        }
        if (bond1 < N_BONDS) {
#if USE_ASYNC_BIN
            float r0 = (float)BinT[(m + 16) * HIDDEN + n0];
            float r1 = (float)BinT[(m + 16) * HIDDEN + n1];
#else
            float r0 = (float)binb[(size_t)bond1 * HIDDEN + n0];
            float r1 = (float)binb[(size_t)bond1 * HIDDEN + n1];
#endif
            msg_out[(size_t)bond1 * HIDDEN + n0] = (bf16_t)fmaxf(c10[r] + r0, 0.f);
            msg_out[(size_t)bond1 * HIDDEN + n1] = (bf16_t)fmaxf(c11[r] + r1, 0.f);
        }
    }
}

// ---------------------------------- atom readout: gather + WMMA(256) + VALU(39) + bias
__global__ void __launch_bounds__(256) k_readout(const bf16_t* __restrict__ msg,
                                                 const int* __restrict__ agraph,
                                                 const bf16_t* __restrict__ Wob,
                                                 const float* __restrict__ fatoms,
                                                 const float* __restrict__ Wo,
                                                 const float* __restrict__ bo,
                                                 float* __restrict__ hidden) {
    __shared__ __align__(16) bf16_t At[M_TILE * A_STRIDE];
    const int tid  = threadIdx.x;
    const int base = blockIdx.x * M_TILE;   // N_ATOMS % 32 == 0: no tail

    {
        int mrow = tid >> 4;
        int ks   = (tid & 15) * 16;
#pragma unroll
        for (int half = 0; half < 2; ++half) {
            int m = mrow + half * 16;
            int atom = base + m;
            float acc[16];
#pragma unroll
            for (int i = 0; i < 16; ++i) acc[i] = 0.f;
            const int* nb = agraph + (size_t)atom * MAX_NB;
#pragma unroll
            for (int j = 0; j < MAX_NB; ++j) {
                int row = nb[j];
                const v8bf* p = (const v8bf*)(msg + (size_t)row * HIDDEN + ks);
                v8bf lo = p[0], hi = p[1];
#pragma unroll
                for (int i = 0; i < 8; ++i) {
                    acc[i]     += (float)lo[i];
                    acc[i + 8] += (float)hi[i];
                }
            }
            v8bf s0, s1;
#pragma unroll
            for (int i = 0; i < 8; ++i) { s0[i] = (bf16_t)acc[i]; s1[i] = (bf16_t)acc[i + 8]; }
            *(v8bf*)&At[m * A_STRIDE + ks]     = s0;
            *(v8bf*)&At[m * A_STRIDE + ks + 8] = s1;
        }
    }
    __syncthreads();

    const int lane = tid & 31;
    const int wave = tid >> 5;
    const int hsel = lane >> 4;
    const int l15  = lane & 15;
    const int n0   = wave * 32 + l15;
    const int n1   = n0 + 16;

    v8f c00 = {}; v8f c01 = {}; v8f c10 = {}; v8f c11 = {};
#pragma unroll
    for (int kc = 0; kc < 8; ++kc) {
        int kb = kc * 32 + hsel * 16;
        v16bf a0 = combine16(*(const v8bf*)&At[l15 * A_STRIDE + kb],
                             *(const v8bf*)&At[l15 * A_STRIDE + kb + 8]);
        v16bf a1 = combine16(*(const v8bf*)&At[(l15 + 16) * A_STRIDE + kb],
                             *(const v8bf*)&At[(l15 + 16) * A_STRIDE + kb + 8]);
        int k0 = kc * 32 + hsel * 16;
        const v8bf* pb0 = (const v8bf*)(Wob + (size_t)n0 * HIDDEN + k0);
        v16bf b0 = combine16(pb0[0], pb0[1]);
        const v8bf* pb1 = (const v8bf*)(Wob + (size_t)n1 * HIDDEN + k0);
        v16bf b1 = combine16(pb1[0], pb1[1]);
        c00 = __builtin_amdgcn_wmma_f32_16x16x32_bf16(false, a0, false, b0, (short)0, c00, false, false);
        c01 = __builtin_amdgcn_wmma_f32_16x16x32_bf16(false, a0, false, b1, (short)0, c01, false, false);
        c10 = __builtin_amdgcn_wmma_f32_16x16x32_bf16(false, a1, false, b0, (short)0, c10, false, false);
        c11 = __builtin_amdgcn_wmma_f32_16x16x32_bf16(false, a1, false, b1, (short)0, c11, false, false);
    }

    const float bias0 = bo[n0];
    const float bias1 = bo[n1];
    const float* w0 = Wo + (size_t)n0 * WO_COLS;  // first 39 columns, fp32
    const float* w1 = Wo + (size_t)n1 * WO_COLS;
#pragma unroll
    for (int r = 0; r < 8; ++r) {
#pragma unroll
        for (int half = 0; half < 2; ++half) {
            int m = hsel * 8 + r + half * 16;
            int atom = base + m;
            float acc0 = (half ? c10[r] : c00[r]) + bias0;
            float acc1 = (half ? c11[r] : c01[r]) + bias1;
            const float* fr = fatoms + (size_t)atom * ATOM_FDIM;
#pragma unroll
            for (int k = 0; k < ATOM_FDIM; ++k) {
                float f = fr[k];
                acc0 = fmaf(f, w0[k], acc0);
                acc1 = fmaf(f, w1[k], acc1);
            }
            hidden[(size_t)atom * HIDDEN + n0] = fmaxf(acc0, 0.f);
            hidden[(size_t)atom * HIDDEN + n1] = fmaxf(acc1, 0.f);
        }
    }
}

// ------------------------------------------------------- deterministic segment mean
__global__ void __launch_bounds__(256) k_pool(const float* __restrict__ hidden,
                                              float* __restrict__ out) {
    int mol = blockIdx.x;
    int h   = threadIdx.x;
    const float* p = hidden + (size_t)mol * APM * HIDDEN + h;
    float s = 0.f;
#pragma unroll 5
    for (int i = 0; i < APM; ++i) s += p[(size_t)i * HIDDEN];
    out[(size_t)mol * HIDDEN + h] = s * (1.0f / (float)APM);
}

// --------------------------------------------------------------------------- launch
extern "C" void kernel_launch(void* const* d_in, const int* in_sizes, int n_in,
                              void* d_out, int out_size, void* d_ws, size_t ws_size,
                              hipStream_t stream) {
    const float* fatoms = (const float*)d_in[0];
    const float* fbonds = (const float*)d_in[1];
    const int*   agraph = (const int*)d_in[2];
    const int*   bgraph = (const int*)d_in[3];
    // d_in[4] = scope (unused: equal contiguous segments)
    const float* W_i = (const float*)d_in[5];
    const float* W_h = (const float*)d_in[6];
    const float* W_o = (const float*)d_in[7];
    const float* b_o = (const float*)d_in[8];
    float* out = (float*)d_out;

    const size_t NBH = (size_t)N_BONDS * HIDDEN;
    bf16_t* msgA = (bf16_t*)d_ws;
    bf16_t* msgB = msgA + NBH;
    bf16_t* binb = msgB + NBH;
    bf16_t* Whb  = binb + NBH;
    bf16_t* Wob  = Whb + (size_t)HIDDEN * HIDDEN;
    float*  hidden = (float*)binb;   // reuse: binput dead after last round

    k_convert<<<256, 256, 0, stream>>>(W_h, W_o, Whb, Wob);
    k_init<<<N_BONDS, 256, 0, stream>>>(fbonds, W_i, binb, msgA);

    bf16_t* in  = msgA;
    bf16_t* outm = msgB;
    const int nblk = (N_BONDS + M_TILE - 1) / M_TILE;
    for (int d = 0; d < DEPTH - 1; ++d) {
        k_round<<<nblk, 256, 0, stream>>>(in, outm, binb, bgraph, Whb);
        bf16_t* t = in; in = outm; outm = t;
    }

    k_readout<<<N_ATOMS / M_TILE, 256, 0, stream>>>(in, agraph, Wob, fatoms, W_o, b_o, hidden);
    k_pool<<<N_MOLS, 256, 0, stream>>>(hidden, out);
}
